// ResnetBlock_78795470012636
// MI455X (gfx1250) — compile-verified, matrix operand order checked by reading, not run on hardware
//
#include <hip/hip_runtime.h>

typedef __attribute__((ext_vector_type(16))) _Float16 v16h;
typedef __attribute__((ext_vector_type(8)))  _Float16 v8h;
typedef __attribute__((ext_vector_type(8)))  float    v8f;

#define B_    4
#define NPTS  8192
#define C0    64
#define CM    128
#define N1    4096
#define N2    2048
#define KNN   16
#define EPS_  1e-5f

static __device__ __forceinline__ v16h cat16(v8h lo, v8h hi) {
  return __builtin_shufflevector(lo, hi, 0,1,2,3,4,5,6,7,8,9,10,11,12,13,14,15);
}
static __device__ __forceinline__ v8f wmma16x16x32(v16h a, v16h b, v8f c) {
  // D = A(16x32 f16) * B(32x16 f16) + C(16x16 f32)
  return __builtin_amdgcn_wmma_f32_16x16x32_f16(false, a, false, b, (short)0, c, false, false);
}

// ---------------------------------------------------------------------------
// Repack an f32 [Krows x 128] weight matrix into f16 WMMA B-fragment order:
// dst[((kc*8 + ct)*32 + lane)*16 + e], K zero-padded to Kpad (multiple of 32).
// Fragment element e of lane L (h = L>>4, n = ct*16 + (L&15)) holds
//   K = kc*32 + ((e&8)<<1) + h*8 + (e&7)     (ISA 16-bit 16x32 striping)
// ---------------------------------------------------------------------------
__global__ void pack_weight(const float* __restrict__ src, _Float16* __restrict__ dst,
                            int Krows, int Kpad) {
  int i = blockIdx.x * 256 + threadIdx.x;
  int total = (Kpad / 32) * 8 * 512;
  if (i >= total) return;
  int e  = i & 15;
  int L  = (i >> 4) & 31;
  int ct = (i >> 9) & 7;
  int kc = i >> 12;
  int h  = L >> 4;
  int n  = ct * 16 + (L & 15);
  int K  = kc * 32 + ((e & 8) << 1) + h * 8 + (e & 7);
  float v = (K < Krows) ? src[(size_t)K * CM + n] : 0.0f;
  dst[i] = (_Float16)v;
}

// ---------------------------------------------------------------------------
// Brute-force KNN: 1 thread per center, LDS-tiled candidate points,
// fully unrolled register top-16 (predicated replace-max, no scratch).
// ---------------------------------------------------------------------------
__global__ __launch_bounds__(256) void knn_kernel(const float* __restrict__ pos,
                                                  int* __restrict__ idxOut,
                                                  int M, int Npts) {
  __shared__ float sx[256], sy[256], sz[256];
  int b = blockIdx.y;
  int c = blockIdx.x * 256 + threadIdx.x;
  const float* P = pos + (size_t)b * NPTS * 3;
  bool active = c < M;
  float cx = 0.f, cy = 0.f, cz = 0.f;
  if (active) { cx = P[c*3+0]; cy = P[c*3+1]; cz = P[c*3+2]; }

  float bd[KNN]; int bi[KNN];
  #pragma unroll
  for (int j = 0; j < KNN; j++) { bd[j] = 3.4e38f; bi[j] = 0; }
  float worst = 3.4e38f;

  for (int base = 0; base < Npts; base += 256) {
    int j = base + threadIdx.x;
    if (j < Npts) {
      sx[threadIdx.x] = P[j*3+0];
      sy[threadIdx.x] = P[j*3+1];
      sz[threadIdx.x] = P[j*3+2];
    }
    __syncthreads();
    int tile = (Npts - base) < 256 ? (Npts - base) : 256;
    if (active) {
      for (int t = 0; t < tile; t++) {
        float dx = sx[t] - cx, dy = sy[t] - cy, dz = sz[t] - cz;
        float d  = dx*dx + dy*dy + dz*dz;
        if (d < worst) {
          int wp = 0; float wv = bd[0];
          #pragma unroll
          for (int q = 1; q < KNN; q++) { if (bd[q] > wv) { wv = bd[q]; wp = q; } }
          #pragma unroll
          for (int q = 0; q < KNN; q++) { if (q == wp) { bd[q] = d; bi[q] = base + t; } }
          worst = bd[0];
          #pragma unroll
          for (int q = 1; q < KNN; q++) worst = fmaxf(worst, bd[q]);
        }
      }
    }
    __syncthreads();
  }
  if (active) {
    #pragma unroll
    for (int j = 0; j < KNN; j++) idxOut[((size_t)b * M + c) * KNN + j] = bi[j];
  }
}

// ---------------------------------------------------------------------------
// DCConv MLP: 8 waves/WG, 8 centers * 16 neighbors = 128 A-rows in LDS.
// Wave w owns row-tile w: GEMM1 (KPAD/32 WMMAs x 8 col-tiles) -> bias+ReLU
// -> f16 LDS -> GEMM2 (4 WMMAs x 8 col-tiles) -> max over 16 rows -> +bias.
// ---------------------------------------------------------------------------
template<int CIN, int KPAD>
__global__ __launch_bounds__(256) void dcconv_kernel(
    const float* __restrict__ pos,        // [B, NPTS, 3]
    const float* __restrict__ feat,       // [B, *, CIN]
    const int*   __restrict__ idx,        // [B, M, 16]
    const _Float16* __restrict__ w0f,     // [KPAD/32][8][32][16]
    const _Float16* __restrict__ w1f,     // [4][8][32][16]
    const float* __restrict__ b0,
    const float* __restrict__ b1,
    float* __restrict__ outRaw,           // [B, M, 128]
    int M, size_t featBatchStride) {
  constexpr int KC1 = KPAD / 32;
  extern __shared__ _Float16 smem[];
  _Float16* Alds = smem;                  // 128 x KPAD
  _Float16* Rlds = smem + 128 * KPAD;     // 128 x 128

  int b   = blockIdx.y;
  int wgc = blockIdx.x * 8;
  int tid = threadIdx.x;
  const float* P = pos  + (size_t)b * NPTS * 3;
  const float* F = feat + (size_t)b * featBatchStride;
  const int*   I = idx  + ((size_t)b * M + wgc) * KNN;

  __builtin_prefetch(w0f, 0, 3);
  __builtin_prefetch(w1f, 0, 3);

  // Build A: gathered neighbor features + relative position + zero pad.
  for (int i = tid; i < 128 * KPAD; i += 256) {
    int row = i / KPAD;
    int k   = i - row * KPAD;
    int lc  = row >> 4;
    int kk  = row & 15;
    int pi  = I[lc * KNN + kk];
    int cen = wgc + lc;
    float v;
    if (k < CIN)          v = F[(size_t)pi * CIN + k];
    else if (k < CIN + 3) v = P[pi * 3 + (k - CIN)] - P[cen * 3 + (k - CIN)];
    else                  v = 0.f;
    Alds[i] = (_Float16)v;
  }
  __syncthreads();

  int wv = tid >> 5;          // wave id == local center == row tile
  int L  = tid & 31;
  int h  = L >> 4;
  int ln = L & 15;

  // A fragments for this row tile (reused across all 8 col tiles)
  v16h afrag[KC1];
  #pragma unroll
  for (int kc = 0; kc < KC1; kc++) {
    const _Float16* ar = &Alds[(wv * 16 + ln) * KPAD + kc * 32 + h * 8];
    afrag[kc] = cat16(*(const v8h*)ar, *(const v8h*)(ar + 16));
  }

  // GEMM1 + bias + ReLU -> Rlds (f16)
  for (int ct = 0; ct < 8; ct++) {
    v8f acc = {0.f,0.f,0.f,0.f,0.f,0.f,0.f,0.f};
    #pragma unroll
    for (int kc = 0; kc < KC1; kc++) {
      const _Float16* wb = &w0f[((size_t)(kc * 8 + ct) * 32 + L) * 16];
      v16h bf = cat16(*(const v8h*)wb, *(const v8h*)(wb + 8));
      acc = wmma16x16x32(afrag[kc], bf, acc);
    }
    float bias = b0[ct * 16 + ln];
    #pragma unroll
    for (int r = 0; r < 8; r++) {
      float v = acc[r] + bias;
      v = v > 0.f ? v : 0.f;
      Rlds[(wv * 16 + h * 8 + r) * CM + ct * 16 + ln] = (_Float16)v;
    }
  }
  __syncthreads();

  // GEMM2 A fragments from ReLU output
  v16h a2[4];
  #pragma unroll
  for (int kc = 0; kc < 4; kc++) {
    const _Float16* ar = &Rlds[(wv * 16 + ln) * CM + kc * 32 + h * 8];
    a2[kc] = cat16(*(const v8h*)ar, *(const v8h*)(ar + 16));
  }

  int cen = wgc + wv;
  for (int ct = 0; ct < 8; ct++) {
    v8f acc = {0.f,0.f,0.f,0.f,0.f,0.f,0.f,0.f};
    #pragma unroll
    for (int kc = 0; kc < 4; kc++) {
      const _Float16* wb = &w1f[((size_t)(kc * 8 + ct) * 32 + L) * 16];
      v16h bf = cat16(*(const v8h*)wb, *(const v8h*)(wb + 8));
      acc = wmma16x16x32(a2[kc], bf, acc);
    }
    // max over the 16 rows (= 16 neighbors of this center)
    float m8 = acc[0];
    #pragma unroll
    for (int r = 1; r < 8; r++) m8 = fmaxf(m8, acc[r]);
    float colmax = fmaxf(m8, __shfl_xor(m8, 16, 32));
    if (h == 0 && cen < M) {
      outRaw[((size_t)b * M + cen) * CM + ct * 16 + ln] = colmax + b1[ct * 16 + ln];
    }
  }
}

// ---------------------------------------------------------------------------
// Residual GEMM: r2 = feat[:, :N2, :64] @ Wres  (WMMA, 128 rows per WG)
// ---------------------------------------------------------------------------
__global__ __launch_bounds__(256) void res_gemm_kernel(
    const float* __restrict__ feat,      // [B, NPTS, C0]
    const _Float16* __restrict__ wrf,    // [2][8][32][16]
    float* __restrict__ out) {           // [B, N2, CM]
  __shared__ _Float16 Alds[128 * C0];
  int b  = blockIdx.y;
  int r0 = blockIdx.x * 128;
  int tid = threadIdx.x;
  const float* F = feat + (size_t)b * NPTS * C0;
  for (int i = tid; i < 128 * C0; i += 256) {
    int row = i >> 6;
    int k   = i & 63;
    Alds[i] = (_Float16)F[(size_t)(r0 + row) * C0 + k];
  }
  __syncthreads();
  int wv = tid >> 5, L = tid & 31, h = L >> 4, ln = L & 15;
  v16h a[2];
  #pragma unroll
  for (int kc = 0; kc < 2; kc++) {
    const _Float16* ar = &Alds[(wv * 16 + ln) * C0 + kc * 32 + h * 8];
    a[kc] = cat16(*(const v8h*)ar, *(const v8h*)(ar + 16));
  }
  for (int ct = 0; ct < 8; ct++) {
    v8f acc = {0.f,0.f,0.f,0.f,0.f,0.f,0.f,0.f};
    #pragma unroll
    for (int kc = 0; kc < 2; kc++) {
      const _Float16* wb = &wrf[((size_t)(kc * 8 + ct) * 32 + L) * 16];
      v16h bf = cat16(*(const v8h*)wb, *(const v8h*)(wb + 8));
      acc = wmma16x16x32(a[kc], bf, acc);
    }
    #pragma unroll
    for (int r = 0; r < 8; r++) {
      int row = r0 + wv * 16 + h * 8 + r;
      out[((size_t)b * N2 + row) * CM + ct * 16 + ln] = acc[r];
    }
  }
}

// ---------------------------------------------------------------------------
// BatchNorm: zero stats, accumulate per-channel sum/sumsq, apply.
// ---------------------------------------------------------------------------
__global__ void bn_init_kernel(float* stats) {
  int i = blockIdx.x * 256 + threadIdx.x;
  if (i < 768) stats[i] = 0.f;
}

__global__ __launch_bounds__(256) void bn_stats_kernel(const float* __restrict__ x,
                                                       float* __restrict__ stats, int rows) {
  int c    = threadIdx.x & 127;
  int half = threadIdx.x >> 7;
  int r0   = blockIdx.x * 128;
  int r1   = r0 + 128 < rows ? r0 + 128 : rows;
  float s = 0.f, s2 = 0.f;
  for (int r = r0 + half; r < r1; r += 2) {
    float v = x[(size_t)r * CM + c];
    s += v; s2 += v * v;
  }
  atomicAdd(&stats[c], s);
  atomicAdd(&stats[128 + c], s2);
}

__global__ void bn_apply_kernel(const float* __restrict__ x, const float* __restrict__ stats,
                                const float* __restrict__ gamma, const float* __restrict__ beta,
                                float* __restrict__ y, int rows) {
  int i = blockIdx.x * 256 + threadIdx.x;
  if (i >= rows * CM) return;
  int c = i & 127;
  float mu  = stats[c] / (float)rows;
  float var = stats[128 + c] / (float)rows - mu * mu;
  y[i] = (x[i] - mu) * rsqrtf(var + EPS_) * gamma[c] + beta[c];
}

__global__ void final_combine_kernel(const float* __restrict__ a, const float* __restrict__ r,
                                     const float* __restrict__ sA, const float* __restrict__ sB,
                                     const float* __restrict__ g2, const float* __restrict__ be2,
                                     const float* __restrict__ gr, const float* __restrict__ br,
                                     float* __restrict__ out, int rows) {
  int i = blockIdx.x * 256 + threadIdx.x;
  if (i >= rows * CM) return;
  int c = i & 127;
  float muA  = sA[c] / (float)rows;
  float varA = sA[128 + c] / (float)rows - muA * muA;
  float muB  = sB[c] / (float)rows;
  float varB = sB[128 + c] / (float)rows - muB * muB;
  float ya = (a[i] - muA) * rsqrtf(varA + EPS_) * g2[c] + be2[c];
  float yb = (r[i] - muB) * rsqrtf(varB + EPS_) * gr[c] + br[c];
  out[i] = ya + yb;
}

__global__ void copy_p2_kernel(const float* __restrict__ pos, float* __restrict__ out) {
  int i = blockIdx.x * 256 + threadIdx.x;
  if (i >= B_ * N2 * 3) return;
  int d = i % 3;
  int rem = i / 3;
  int n = rem % N2;
  int b = rem / N2;
  out[i] = pos[((size_t)b * NPTS + n) * 3 + d];
}

// ---------------------------------------------------------------------------
extern "C" void kernel_launch(void* const* d_in, const int* in_sizes, int n_in,
                              void* d_out, int out_size, void* d_ws, size_t ws_size,
                              hipStream_t stream) {
  const float* pos    = (const float*)d_in[0];
  const float* feat   = (const float*)d_in[1];
  const float* W0_1   = (const float*)d_in[2];
  const float* b0_1   = (const float*)d_in[3];
  const float* W1_1   = (const float*)d_in[4];
  const float* b1_1   = (const float*)d_in[5];
  const float* Wres   = (const float*)d_in[6];
  const float* gamma1 = (const float*)d_in[7];
  const float* beta1  = (const float*)d_in[8];
  const float* W0_2   = (const float*)d_in[9];
  const float* b0_2   = (const float*)d_in[10];
  const float* W1_2   = (const float*)d_in[11];
  const float* b1_2   = (const float*)d_in[12];
  const float* gamma2 = (const float*)d_in[13];
  const float* beta2  = (const float*)d_in[14];
  const float* gammar = (const float*)d_in[15];
  const float* betar  = (const float*)d_in[16];
  (void)in_sizes; (void)n_in; (void)out_size; (void)ws_size;

  char* ws = (char*)d_ws;
  size_t off = 0;
  auto alloc = [&](size_t bytes) -> char* {
    off = (off + 255) & ~(size_t)255;
    char* p = ws + off;
    off += bytes;
    return p;
  };
  _Float16* wf0 = (_Float16*)alloc((size_t)96  * 128 * 2);
  _Float16* wf1 = (_Float16*)alloc((size_t)128 * 128 * 2);
  _Float16* wf2 = (_Float16*)alloc((size_t)160 * 128 * 2);
  _Float16* wf3 = (_Float16*)alloc((size_t)128 * 128 * 2);
  _Float16* wfr = (_Float16*)alloc((size_t)64  * 128 * 2);
  int*   idx1  = (int*)alloc((size_t)B_ * N1 * KNN * 4);
  int*   idx2  = (int*)alloc((size_t)B_ * N2 * KNN * 4);
  float* c1raw = (float*)alloc((size_t)B_ * N1 * CM * 4);
  float* c1    = (float*)alloc((size_t)B_ * N1 * CM * 4);
  float* c2raw = (float*)alloc((size_t)B_ * N2 * CM * 4);
  float* r2    = (float*)alloc((size_t)B_ * N2 * CM * 4);
  float* stats = (float*)alloc(768 * 4);   // [0:256)=c2, [256:512)=r2, [512:768)=c1
  float* statsC2 = stats;
  float* statsR2 = stats + 256;
  float* statsC1 = stats + 512;

  // 1. Pack weights into WMMA fragment order (f16).
  pack_weight<<<(3*8*512 + 255)/256, 256, 0, stream>>>(W0_1, wf0, 67, 96);
  pack_weight<<<(4*8*512 + 255)/256, 256, 0, stream>>>(W1_1, wf1, 128, 128);
  pack_weight<<<(5*8*512 + 255)/256, 256, 0, stream>>>(W0_2, wf2, 131, 160);
  pack_weight<<<(4*8*512 + 255)/256, 256, 0, stream>>>(W1_2, wf3, 128, 128);
  pack_weight<<<(2*8*512 + 255)/256, 256, 0, stream>>>(Wres, wfr, 64, 64);

  // 2. KNN for both levels (centers are prefixes of pos).
  knn_kernel<<<dim3(N1/256, B_), 256, 0, stream>>>(pos, idx1, N1, NPTS);
  knn_kernel<<<dim3(N2/256, B_), 256, 0, stream>>>(pos, idx2, N2, N1);

  // 3. DCConv layer 1 (Cin=64 -> A row 67, pad 96).
  size_t sh1 = (size_t)(128*96 + 128*128) * sizeof(_Float16);
  dcconv_kernel<64,96><<<dim3(N1/8, B_), 256, sh1, stream>>>(
      pos, feat, idx1, wf0, wf1, b0_1, b1_1, c1raw, N1, (size_t)NPTS * C0);

  // 4. BN1 -> c1.
  bn_init_kernel<<<3, 256, 0, stream>>>(stats);
  bn_stats_kernel<<<(B_*N1)/128, 256, 0, stream>>>(c1raw, statsC1, B_*N1);
  bn_apply_kernel<<<(B_*N1*CM + 255)/256, 256, 0, stream>>>(c1raw, statsC1, gamma1, beta1, c1, B_*N1);

  // 5. DCConv layer 2 (Cin=128 -> A row 131, pad 160).
  size_t sh2 = (size_t)(128*160 + 128*128) * sizeof(_Float16);
  dcconv_kernel<128,160><<<dim3(N2/8, B_), 256, sh2, stream>>>(
      pos, c1, idx2, wf2, wf3, b0_2, b1_2, c2raw, N2, (size_t)N1 * CM);

  // 6. Residual path: r2 = feat[:, :2048] @ Wres.
  res_gemm_kernel<<<dim3(N2/128, B_), 256, 0, stream>>>(feat, wfr, r2);

  // 7. BN stats for c2raw and r2, then fused combine into d_out.
  bn_stats_kernel<<<(B_*N2)/128, 256, 0, stream>>>(c2raw, statsC2, B_*N2);
  bn_stats_kernel<<<(B_*N2)/128, 256, 0, stream>>>(r2, statsR2, B_*N2);

  float* outPos = (float*)d_out;                 // p2: [B, N2, 3]
  float* outC   = outPos + (size_t)B_ * N2 * 3;  // c2: [B, N2, 128]
  copy_p2_kernel<<<(B_*N2*3 + 255)/256, 256, 0, stream>>>(pos, outPos);
  final_combine_kernel<<<(B_*N2*CM + 255)/256, 256, 0, stream>>>(
      c2raw, r2, statsC2, statsR2, gamma2, beta2, gammar, betar, outC, B_*N2);
}
